// GCNModel_6090263626385
// MI455X (gfx1250) — matrix operand order repaired and checked by reading, not compiled, not run on hardware
//
#include <hip/hip_runtime.h>
#include <hip/hip_bf16.h>

// ---------------------------------------------------------------------------
// GCN (2-layer, PyG GCNConv semantics) for MI455X / gfx1250, wave32.
// GEMMs via v_wmma_f32_16x16x32_f16 (f16 in, f32 accumulate), weights kept
// transposed so every B fragment is one contiguous 32B per-lane load.
// Edge aggregation via L2-resident global_atomic_add_f32 scatter.
// ---------------------------------------------------------------------------

typedef __attribute__((ext_vector_type(16))) _Float16 v16h;
typedef __attribute__((ext_vector_type(8)))  _Float16 v8h;
typedef __attribute__((ext_vector_type(8)))  float    v8f;

#define IN_DIM  128
#define HID_DIM 64
#define OUT_DIM 40
#define OUT_PAD 48   // pad 40 -> 48 so every WMMA N-tile is full

// ---------------- WMMA fragment loaders (CDNA5 wave32 layouts) -------------

// A fragment: 16x32 f16 tile of row-major [M x K] matrix (lda = K).
// lane L: row m = L&15; halves h=0..15 -> K = kbase+h
// where kbase = (L>=16) ? 16 : 0. Two contiguous 16B loads.
__device__ __forceinline__ v16h load_a_frag(const _Float16* A, int row0, int lda,
                                            int k0, int lane) {
    int m     = lane & 15;
    int kbase = (lane >> 4) ? 16 : 0;
    const _Float16* p = A + (size_t)(row0 + m) * lda + k0 + kbase;
    v8h lo = *(const v8h*)(p);
    v8h hi = *(const v8h*)(p + 8);
    v16h a;
#pragma unroll
    for (int h = 0; h < 8; ++h) { a[h] = lo[h]; a[8 + h] = hi[h]; }
    return a;
}

// B fragment from a TRANSPOSED weight matrix Bt stored [N x K] (ldt = K).
// lane L: col n = col0+(L&15); halves h=0..15 -> K = k0 + 16*(L>=16) + h,
// i.e. 16 contiguous halves = one aligned 32B load per lane.
__device__ __forceinline__ v16h load_bt_frag(const _Float16* Bt, int col0, int ldt,
                                             int k0, int lane) {
    int n  = col0 + (lane & 15);
    int kb = k0 + ((lane >> 4) ? 16 : 0);
    return *(const v16h*)(Bt + (size_t)n * ldt + kb);
}

// ---------------- conversion / prep kernels --------------------------------

__global__ void k_f32_to_f16(const float* __restrict__ src,
                             _Float16* __restrict__ dst, int n) {
    int i = blockIdx.x * blockDim.x + threadIdx.x;
    if (i < n) dst[i] = (_Float16)src[i];
}

// w1t[n*IN_DIM + k] = f16(W1[k*HID_DIM + n])   (HID_DIM x IN_DIM)
__global__ void k_conv_w1t(const float* __restrict__ w1,
                           _Float16* __restrict__ w1t) {
    int i = blockIdx.x * blockDim.x + threadIdx.x;
    if (i >= HID_DIM * IN_DIM) return;
    int n = i / IN_DIM, k = i % IN_DIM;
    w1t[i] = (_Float16)w1[k * HID_DIM + n];
}

// w2t[n*HID_DIM + k] = f16(W2[k*OUT_DIM + n]) for n<40, else 0  (OUT_PAD x HID_DIM)
__global__ void k_conv_w2t(const float* __restrict__ w2,
                           _Float16* __restrict__ w2t) {
    int i = blockIdx.x * blockDim.x + threadIdx.x;
    if (i >= OUT_PAD * HID_DIM) return;
    int n = i / HID_DIM, k = i % HID_DIM;
    w2t[i] = (n < OUT_DIM) ? (_Float16)w2[k * OUT_DIM + n] : (_Float16)0.0f;
}

__global__ void k_degree(const long long* __restrict__ dst_idx,
                         unsigned* __restrict__ deg, int E) {
    int e = blockIdx.x * blockDim.x + threadIdx.x;
    if (e < E) atomicAdd(&deg[(int)dst_idx[e]], 1u);
}

__global__ void k_dis(const unsigned* __restrict__ deg,
                      float* __restrict__ dis, int N) {
    int i = blockIdx.x * blockDim.x + threadIdx.x;
    if (i < N) dis[i] = rsqrtf((float)(deg[i] + 1u));   // +1 = self-loop
}

// int64 -> int32 indices + per-edge symmetric norm
__global__ void k_edge_prep(const long long* __restrict__ ei,
                            const float* __restrict__ dis,
                            int* __restrict__ s32, int* __restrict__ d32,
                            float* __restrict__ enorm, int E) {
    int e = blockIdx.x * blockDim.x + threadIdx.x;
    if (e >= E) return;
    int s = (int)ei[e];
    int d = (int)ei[(size_t)e + (size_t)E];
    s32[e] = s; d32[e] = d;
    enorm[e] = dis[s] * dis[d];
}

// ---------------- WMMA GEMMs (32 rows per wave, B shared by 2 A-tiles) -----

// xw1[N,64] = xh[N,128] @ W1; w1t is W1 transposed [64 x 128] f16.
__global__ void k_gemm1(const _Float16* __restrict__ xh,
                        const _Float16* __restrict__ w1t,
                        float* __restrict__ xw1, int nrows) {
    int wave = (int)((blockIdx.x * blockDim.x + threadIdx.x) >> 5);
    int lane = threadIdx.x & 31;
    int row0 = wave * 32;
    if (row0 >= nrows) return;
    v8f acc0[4] = {v8f{}, v8f{}, v8f{}, v8f{}};
    v8f acc1[4] = {v8f{}, v8f{}, v8f{}, v8f{}};
#pragma unroll
    for (int kk = 0; kk < IN_DIM; kk += 32) {
        v16h a0 = load_a_frag(xh, row0,      IN_DIM, kk, lane);
        v16h a1 = load_a_frag(xh, row0 + 16, IN_DIM, kk, lane);
#pragma unroll
        for (int t = 0; t < 4; ++t) {
            v16h b = load_bt_frag(w1t, t * 16, IN_DIM, kk, lane);
            acc0[t] = __builtin_amdgcn_wmma_f32_16x16x32_f16(
                false, a0, false, b, (short)0, acc0[t], false, false);
            acc1[t] = __builtin_amdgcn_wmma_f32_16x16x32_f16(
                false, a1, false, b, (short)0, acc1[t], false, false);
        }
    }
    int m = (lane >> 4) ? 8 : 0;
    int n = lane & 15;
#pragma unroll
    for (int t = 0; t < 4; ++t)
#pragma unroll
        for (int r = 0; r < 8; ++r) {
            xw1[(size_t)(row0 + m + r) * HID_DIM + t * 16 + n]      = acc0[t][r];
            xw1[(size_t)(row0 + 16 + m + r) * HID_DIM + t * 16 + n] = acc1[t][r];
        }
}

// xw2[N,48] = h1h[N,64] @ W2; w2t is padded W2 transposed [48 x 64] f16.
__global__ void k_gemm2(const _Float16* __restrict__ h1h,
                        const _Float16* __restrict__ w2t,
                        float* __restrict__ xw2, int nrows) {
    int wave = (int)((blockIdx.x * blockDim.x + threadIdx.x) >> 5);
    int lane = threadIdx.x & 31;
    int row0 = wave * 32;
    if (row0 >= nrows) return;
    v8f acc0[3] = {v8f{}, v8f{}, v8f{}};
    v8f acc1[3] = {v8f{}, v8f{}, v8f{}};
#pragma unroll
    for (int kk = 0; kk < HID_DIM; kk += 32) {
        v16h a0 = load_a_frag(h1h, row0,      HID_DIM, kk, lane);
        v16h a1 = load_a_frag(h1h, row0 + 16, HID_DIM, kk, lane);
#pragma unroll
        for (int t = 0; t < 3; ++t) {
            v16h b = load_bt_frag(w2t, t * 16, HID_DIM, kk, lane);
            acc0[t] = __builtin_amdgcn_wmma_f32_16x16x32_f16(
                false, a0, false, b, (short)0, acc0[t], false, false);
            acc1[t] = __builtin_amdgcn_wmma_f32_16x16x32_f16(
                false, a1, false, b, (short)0, acc1[t], false, false);
        }
    }
    int m = (lane >> 4) ? 8 : 0;
    int n = lane & 15;
#pragma unroll
    for (int t = 0; t < 3; ++t)
#pragma unroll
        for (int r = 0; r < 8; ++r) {
            xw2[(size_t)(row0 + m + r) * OUT_PAD + t * 16 + n]      = acc0[t][r];
            xw2[(size_t)(row0 + 16 + m + r) * OUT_PAD + t * 16 + n] = acc1[t][r];
        }
}

// ---------------- edge scatter (message passing) ---------------------------

// layer 1: 64 threads per edge; agg1[dst,f] += xw1[src,f] * norm[e]
__global__ void k_scatter1(const int* __restrict__ s32, const int* __restrict__ d32,
                           const float* __restrict__ enorm,
                           const float* __restrict__ xw1,
                           float* __restrict__ agg1, int E) {
    int idx = blockIdx.x * blockDim.x + threadIdx.x;
    int e = idx >> 6, f = idx & 63;
    if (e >= E) return;
    int s = s32[e], d = d32[e];
    float v = xw1[(size_t)s * HID_DIM + f] * enorm[e];
    atomicAdd(&agg1[(size_t)d * HID_DIM + f], v);
}

// h1h = f16(relu(agg1 + dis^2 * xw1 + b1))  (dis^2*xw1 = self-loop message)
__global__ void k_bias_relu_f16(const float* __restrict__ agg1,
                                const float* __restrict__ xw1,
                                const float* __restrict__ dis,
                                const float* __restrict__ b1,
                                _Float16* __restrict__ h1h, int total) {
    int i = blockIdx.x * blockDim.x + threadIdx.x;
    if (i >= total) return;
    int node = i >> 6, f = i & 63;
    float dd = dis[node]; dd *= dd;
    float v = agg1[i] + dd * xw1[i] + b1[f];
    h1h[i] = (_Float16)fmaxf(v, 0.0f);
}

// layer 2: 40 threads per edge; agg2[dst,f] += xw2[src,f] * norm[e]
__global__ void k_scatter2(const int* __restrict__ s32, const int* __restrict__ d32,
                           const float* __restrict__ enorm,
                           const float* __restrict__ xw2,
                           float* __restrict__ agg2, int E) {
    int idx = blockIdx.x * blockDim.x + threadIdx.x;
    int e = idx / OUT_DIM, f = idx % OUT_DIM;
    if (e >= E) return;
    int s = s32[e], d = d32[e];
    float v = xw2[(size_t)s * OUT_PAD + f] * enorm[e];
    atomicAdd(&agg2[(size_t)d * OUT_DIM + f], v);
}

// out = log_softmax(agg2 + dis^2*xw2 + b2); one wave per node (wave32).
__global__ void k_finalize(const float* __restrict__ agg2,
                           const float* __restrict__ xw2,
                           const float* __restrict__ dis,
                           const float* __restrict__ b2,
                           float* __restrict__ out, int N) {
    int wave = (int)((blockIdx.x * blockDim.x + threadIdx.x) >> 5);
    int lane = threadIdx.x & 31;
    if (wave >= N) return;
    float dd = dis[wave]; dd *= dd;
    float z0 = agg2[(size_t)wave * OUT_DIM + lane]
             + dd * xw2[(size_t)wave * OUT_PAD + lane] + b2[lane];
    float z1 = -3.0e38f;
    if (lane < OUT_DIM - 32)
        z1 = agg2[(size_t)wave * OUT_DIM + 32 + lane]
           + dd * xw2[(size_t)wave * OUT_PAD + 32 + lane] + b2[32 + lane];
    float m = fmaxf(z0, z1);
#pragma unroll
    for (int off = 16; off > 0; off >>= 1) m = fmaxf(m, __shfl_xor(m, off));
    float s = __expf(z0 - m) + ((lane < OUT_DIM - 32) ? __expf(z1 - m) : 0.0f);
#pragma unroll
    for (int off = 16; off > 0; off >>= 1) s += __shfl_xor(s, off);
    float lse = m + __logf(s);
    out[(size_t)wave * OUT_DIM + lane] = z0 - lse;
    if (lane < OUT_DIM - 32)
        out[(size_t)wave * OUT_DIM + 32 + lane] = z1 - lse;
}

// ---------------------------------------------------------------------------

extern "C" void kernel_launch(void* const* d_in, const int* in_sizes, int n_in,
                              void* d_out, int out_size, void* d_ws, size_t ws_size,
                              hipStream_t stream) {
    const float*     x  = (const float*)d_in[0];
    const long long* ei = (const long long*)d_in[1];   // int64 [2, E]
    const float*     W1 = (const float*)d_in[2];
    const float*     b1 = (const float*)d_in[3];
    const float*     W2 = (const float*)d_in[4];
    const float*     b2 = (const float*)d_in[5];
    float*           out = (float*)d_out;

    const int N = in_sizes[0] / IN_DIM;
    const int E = in_sizes[1] / 2;

    // ---- workspace carve-up (256B aligned), with buffer reuse ----
    size_t off = 0;
    auto carve = [&](size_t bytes) -> char* {
        char* p = (char*)d_ws + off;
        off += (bytes + 255) & ~(size_t)255;
        return p;
    };
    unsigned* deg   = (unsigned*)carve((size_t)N * 4);
    float*    dis   = (float*)   carve((size_t)N * 4);
    int*      s32   = (int*)     carve((size_t)E * 4);
    int*      d32   = (int*)     carve((size_t)E * 4);
    float*    enorm = (float*)   carve((size_t)E * 4);
    _Float16* w1t   = (_Float16*)carve((size_t)HID_DIM * IN_DIM * 2);
    _Float16* w2t   = (_Float16*)carve((size_t)OUT_PAD * HID_DIM * 2);
    char*     bufA  = carve((size_t)N * IN_DIM * 2);   // xh, later h1h
    char*     bufB  = carve((size_t)N * HID_DIM * 4);  // xw1, later agg2
    char*     bufC  = carve((size_t)N * HID_DIM * 4);  // agg1, later xw2

    _Float16* xh   = (_Float16*)bufA;
    float*    xw1  = (float*)bufB;
    float*    agg1 = (float*)bufC;
    _Float16* h1h  = (_Float16*)bufA;   // reuse (x dead after gemm1)
    float*    xw2  = (float*)bufC;      // reuse (agg1 dead after bias_relu)
    float*    agg2 = (float*)bufB;      // reuse (xw1 dead after bias_relu)

    const int B256 = 256;

    // ---- degrees & normalization ----
    hipMemsetAsync(deg, 0, (size_t)N * 4, stream);
    k_degree<<<(E + B256 - 1) / B256, B256, 0, stream>>>(ei + (size_t)E, deg, E);
    k_dis<<<(N + B256 - 1) / B256, B256, 0, stream>>>(deg, dis, N);
    k_edge_prep<<<(E + B256 - 1) / B256, B256, 0, stream>>>(ei, dis, s32, d32, enorm, E);

    // ---- f16 conversions (weights transposed for WMMA B layout) ----
    k_f32_to_f16<<<((size_t)N * IN_DIM + B256 - 1) / B256, B256, 0, stream>>>(x, xh, N * IN_DIM);
    k_conv_w1t<<<(HID_DIM * IN_DIM + B256 - 1) / B256, B256, 0, stream>>>(W1, w1t);
    k_conv_w2t<<<(OUT_PAD * HID_DIM + B256 - 1) / B256, B256, 0, stream>>>(W2, w2t);

    // ---- layer 1: GEMM (WMMA) + scatter + bias/relu ----
    {
        int waves = (N + 31) / 32;                         // one wave per 32 rows
        int blocks = (waves + 3) / 4;                      // 128 thr = 4 waves
        k_gemm1<<<blocks, 128, 0, stream>>>(xh, w1t, xw1, N);
    }
    hipMemsetAsync(agg1, 0, (size_t)N * HID_DIM * 4, stream);
    {
        long long total = (long long)E * HID_DIM;
        k_scatter1<<<(unsigned)((total + B256 - 1) / B256), B256, 0, stream>>>(
            s32, d32, enorm, xw1, agg1, E);
    }
    k_bias_relu_f16<<<((size_t)N * HID_DIM + B256 - 1) / B256, B256, 0, stream>>>(
        agg1, xw1, dis, b1, h1h, N * HID_DIM);

    // ---- layer 2: GEMM (WMMA) + scatter + log_softmax ----
    hipMemsetAsync(agg2, 0, (size_t)N * OUT_DIM * 4, stream);
    {
        int waves = (N + 31) / 32;
        int blocks = (waves + 3) / 4;
        k_gemm2<<<blocks, 128, 0, stream>>>(h1h, w2t, xw2, N);
    }
    {
        long long total = (long long)E * OUT_DIM;
        k_scatter2<<<(unsigned)((total + B256 - 1) / B256), B256, 0, stream>>>(
            s32, d32, enorm, xw2, agg2, E);
    }
    {
        int blocks = (N + 7) / 8;                          // 8 waves/block, 1 node/wave
        k_finalize<<<blocks, 256, 0, stream>>>(agg2, xw2, dis, b2, out, N);
    }
}
